// AttentionHead_19344532701481
// MI455X (gfx1250) — matrix operand order repaired and checked by reading, not compiled, not run on hardware
//
#include <hip/hip_runtime.h>
#include <hip/hip_bf16.h>

// ---------------------------------------------------------------------------
// AttentionHead on MI455X (gfx1250): bf16 WMMA flash-attention with
// double-buffered GLOBAL_LOAD_ASYNC_TO_LDS staging of K/V tiles.
//   B=4, N=4096, D=1024, H=128.  Q/K/V staged as bf16 in d_ws, V transposed.
// ---------------------------------------------------------------------------

typedef __attribute__((ext_vector_type(16))) __bf16 v16bf;
typedef __attribute__((ext_vector_type(8)))  __bf16 v8bf;
typedef __attribute__((ext_vector_type(8)))  float  v8f;
typedef __attribute__((ext_vector_type(8)))  float  f32x8;

union V16 { v16bf v; v8bf h[2]; };

static __device__ __forceinline__ v8f wmma_bf16(v16bf a, v16bf b, v8f c) {
    return __builtin_amdgcn_wmma_f32_16x16x32_bf16(
        /*neg_a=*/false, a, /*neg_b=*/false, b,
        /*c_mod=*/(short)0, c, /*reuse_a=*/false, /*reuse_b=*/false);
}

#define NSEQ   4096
#define DMODEL 1024
#define DHEAD  128

// ---------------------------------------------------------------------------
// W [D][H] fp32 (row-major)  ->  Wt [H][D] bf16  (so B-fragments are contiguous)
// ---------------------------------------------------------------------------
__global__ __launch_bounds__(128) void wt_convert(const float* __restrict__ W,
                                                  __bf16* __restrict__ Wt) {
    int idx = blockIdx.x * 128 + threadIdx.x;          // 1024*128 elements
    int k = idx >> 7;                                  // 0..1023
    int h = idx & 127;                                 // 0..127 (coalesced read)
    Wt[(long)h * DMODEL + k] = (__bf16)W[(long)k * DHEAD + h];
}

// ---------------------------------------------------------------------------
// proj_gemm: Out[r][c] = scale * sum_k X[r][k] * Wt[c][k]
//   block = 128 threads = 4 waves; each wave: 16 rows x 128 cols (8 accums)
//   A-frag per-lane K slots: {half*16+0..15}               (matches B map)
//   B-frag per-lane K slots: {half*16+0..15}               (contiguous, same as A)
//   transposeOut: store as [B][H][N] instead of [B*N][H] (used for V).
// ---------------------------------------------------------------------------
__global__ __launch_bounds__(128) void proj_gemm(const float*  __restrict__ X,
                                                 const __bf16* __restrict__ Wt,
                                                 __bf16*       __restrict__ Out,
                                                 float scale, int transposeOut) {
    const int lane = threadIdx.x & 31;
    const int wave = threadIdx.x >> 5;
    const int half = lane >> 4;
    const int l16  = lane & 15;
    const long rowBase = (long)blockIdx.x * 64 + wave * 16;

    v8f acc[8] = {};

    for (int k0 = 0; k0 < DMODEL; k0 += 32) {
        // ---- A fragment: 16 rows x 32 k of X, fp32 -> bf16 --------------
        const float* xa = X + (rowBase + l16) * (long)DMODEL + k0 + half * 16;
        f32x8 x0 = *(const f32x8*)xa;          // k slots 0..7 of this lane
        f32x8 x1 = *(const f32x8*)(xa + 8);    // k slots 8..15 of this lane
        V16 a;
        v8bf b0, b1;
#pragma unroll
        for (int i = 0; i < 8; ++i) { b0[i] = (__bf16)x0[i]; b1[i] = (__bf16)x1[i]; }
        a.h[0] = b0; a.h[1] = b1;

        // ---- 8 B fragments (16 cols each) + WMMA ------------------------
#pragma unroll
        for (int j = 0; j < 8; ++j) {
            const __bf16* wp = Wt + (long)(j * 16 + l16) * DMODEL + k0 + half * 16;
            v16bf bf = *(const v16bf*)wp;      // 32B contiguous, 32B aligned
            acc[j] = wmma_bf16(a.v, bf, acc[j]);
        }
    }

    // ---- store (C/D layout: col = l16, row = vgpr + 8*half) -------------
#pragma unroll
    for (int j = 0; j < 8; ++j) {
#pragma unroll
        for (int v = 0; v < 8; ++v) {
            long r  = rowBase + v + 8 * half;
            int  c  = j * 16 + l16;
            float val = acc[j][v] * scale;
            if (!transposeOut) {
                Out[r * DHEAD + c] = (__bf16)val;
            } else {
                long b = r >> 12;              // r / 4096
                long n = r & (NSEQ - 1);
                Out[(b * DHEAD + c) * (long)NSEQ + n] = (__bf16)val;
            }
        }
    }
}

// ---------------------------------------------------------------------------
// Async staging helpers: per-lane 16B transfers, IOFFSET applies to BOTH the
// LDS and the global side (ISA 10.7 pseudocode), so contiguous blocks stage
// with one base address + immediate offsets.  Tracked by ASYNCcnt.
// ---------------------------------------------------------------------------
static __device__ __forceinline__ void async_copy_16(unsigned lds, unsigned long long g) {
    asm volatile("global_load_async_to_lds_b128 %0, %1, off"
                 :: "v"(lds), "v"(g) : "memory");
}
static __device__ __forceinline__ void wait_async0() {
    asm volatile("s_wait_asynccnt 0x0" ::: "memory");
}

// ---------------------------------------------------------------------------
// flash_attn: out = softmax(Q K^T) V, online softmax over 32-key tiles.
//   grid = (N/64, B); block = 128 threads = 4 waves; wave owns 16 q-rows.
//   Qb, Kb : [B*N][128] bf16 (Q pre-scaled by 1/sqrt(1024))
//   Vt     : [B][128][N]  bf16 (transposed so P*V B-frags are contiguous)
//   K/V tiles staged once per workgroup into LDS via async loads, double
//   buffered so tile i+1's copy overlaps tile i's WMMAs + softmax.
// ---------------------------------------------------------------------------
__global__ __launch_bounds__(128) void flash_attn(const __bf16* __restrict__ Qb,
                                                  const __bf16* __restrict__ Kb,
                                                  const __bf16* __restrict__ Vt,
                                                  float* __restrict__ Out) {
    __shared__ __align__(128) __bf16 KT[2][32][128];   // 2 x 8 KB key tiles
    __shared__ __align__(128) __bf16 VT[2][128][32];   // 2 x 8 KB value tiles
    __shared__ __align__(32)  __bf16 Pbuf[4][16][32];  // per-wave P tile (4 KB)

    const int  tid  = threadIdx.x;
    const int  lane = tid & 31;
    const int  wave = tid >> 5;
    const int  half = lane >> 4;
    const int  l16  = lane & 15;
    const int  b    = blockIdx.y;
    const long q0   = (long)blockIdx.x * 64 + wave * 16;

    const __bf16* Qbase = Qb + (long)b * NSEQ * DHEAD;
    const __bf16* Kbase = Kb + (long)b * NSEQ * DHEAD;
    const __bf16* Vbase = Vt + (long)b * DHEAD * NSEQ;

    // ---- cooperative async stage of one 32-key K tile + V tile ----------
    auto stage_tiles = [&](int buf, int kv) {
        // K tile: rows kv..kv+31 of [N][128] bf16 are 8192 contiguous bytes.
        unsigned klds = (unsigned)(size_t)&KT[buf][0][0] + tid * 16;
        unsigned long long kg = (unsigned long long)(const void*)Kbase
                              + (unsigned long long)kv * 256ull + tid * 16;
        async_copy_16(klds,        kg);
        async_copy_16(klds + 2048, kg + 2048);
        async_copy_16(klds + 4096, kg + 4096);
        async_copy_16(klds + 6144, kg + 6144);
        // V tile: 128 rows (h) x 64 B each from [H][N] bf16 (row stride 8192 B).
        unsigned vlds = (unsigned)(size_t)&VT[buf][0][0] + tid * 64;
        unsigned long long vg = (unsigned long long)(const void*)Vbase
                              + (unsigned long long)tid * 8192ull
                              + (unsigned long long)kv * 2ull;
        async_copy_16(vlds,      vg);
        async_copy_16(vlds + 16, vg + 16);
        async_copy_16(vlds + 32, vg + 32);
        async_copy_16(vlds + 48, vg + 48);
    };

    // ---- Q fragments: 4 chunks of 16x32 (whole head dim), kept resident --
    V16 qf[4];
#pragma unroll
    for (int hc = 0; hc < 4; ++hc) {
        const __bf16* qp = Qbase + (q0 + l16) * DHEAD + hc * 32 + half * 16;
        qf[hc].h[0] = *(const v8bf*)qp;
        qf[hc].h[1] = *(const v8bf*)(qp + 8);
    }

    v8f   o[8] = {};
    float m_i[8], l_i[8];
#pragma unroll
    for (int j = 0; j < 8; ++j) { m_i[j] = -1e30f; l_i[j] = 0.0f; }

    stage_tiles(0, 0);                       // prologue: tile 0 in flight

    for (int kv = 0; kv < NSEQ; kv += 32) {
        const int cur = (kv >> 5) & 1;
        wait_async0();                       // this wave's async copies done
        __syncthreads();                     // all waves' copies visible; prior
                                             // reads of buffer `cur` finished
        if (kv + 32 < NSEQ) stage_tiles(cur ^ 1, kv + 32);

        // ---- S = Q K^T : 2 accums of 16x16 (32 keys), operands from LDS -
        v8f s[2] = {};
#pragma unroll
        for (int g = 0; g < 2; ++g) {
#pragma unroll
            for (int hc = 0; hc < 4; ++hc) {
                v16bf kf = *(const v16bf*)&KT[cur][g * 16 + l16][hc * 32 + half * 16];
                s[g] = wmma_bf16(qf[hc].v, kf, s[g]);
            }
        }

        // ---- online softmax (row r = j + 8*half, key col = l16) ---------
        float alpha[8];
#pragma unroll
        for (int j = 0; j < 8; ++j) {
            float t = fmaxf(s[0][j], s[1][j]);
#pragma unroll
            for (int msk = 1; msk < 16; msk <<= 1)
                t = fmaxf(t, __shfl_xor(t, msk, 16));
            float mn = fmaxf(m_i[j], t);
            alpha[j] = __expf(m_i[j] - mn);
            m_i[j]   = mn;
            float p0 = __expf(s[0][j] - mn);
            float p1 = __expf(s[1][j] - mn);
            Pbuf[wave][j + 8 * half][l16]      = (__bf16)p0;
            Pbuf[wave][j + 8 * half][16 + l16] = (__bf16)p1;
            float rs = p0 + p1;
#pragma unroll
            for (int msk = 1; msk < 16; msk <<= 1)
                rs += __shfl_xor(rs, msk, 16);
            l_i[j] = l_i[j] * alpha[j] + rs;
#pragma unroll
            for (int a = 0; a < 8; ++a) o[a][j] *= alpha[j];
        }
        __syncthreads();

        // ---- P as A-fragment from LDS (per-lane K slots half*16+0..15) ----
        V16 pf;
        const __bf16* pp = &Pbuf[wave][l16][half * 16];
        pf.h[0] = *(const v8bf*)pp;
        pf.h[1] = *(const v8bf*)(pp + 8);

        // ---- O += P * V : 8 col-blocks of 16, V operand from LDS --------
#pragma unroll
        for (int j = 0; j < 8; ++j) {
            v16bf vf = *(const v16bf*)&VT[cur][j * 16 + l16][half * 16];
            o[j] = wmma_bf16(pf.v, vf, o[j]);
        }
        __syncthreads();
    }

    // ---- epilogue: O /= l, store fp32 -----------------------------------
    float inv[8];
#pragma unroll
    for (int j = 0; j < 8; ++j) inv[j] = (l_i[j] > 0.0f) ? 1.0f / l_i[j] : 0.0f;
#pragma unroll
    for (int a = 0; a < 8; ++a) {
#pragma unroll
        for (int j = 0; j < 8; ++j) {
            long r = q0 + j + 8 * half;
            Out[((long)b * NSEQ + r) * DHEAD + a * 16 + l16] = o[a][j] * inv[j];
        }
    }
}

// ---------------------------------------------------------------------------
extern "C" void kernel_launch(void* const* d_in, const int* in_sizes, int n_in,
                              void* d_out, int out_size, void* d_ws, size_t ws_size,
                              hipStream_t stream) {
    const float* X  = (const float*)d_in[0];
    const float* Wq = (const float*)d_in[1];
    const float* Wk = (const float*)d_in[2];
    const float* Wv = (const float*)d_in[3];

    const long BN = 4L * NSEQ;                     // 16384 rows
    char* ws = (char*)d_ws;
    size_t off = 0;
    auto take = [&](size_t bytes) { char* p = ws + off; off = (off + bytes + 255) & ~(size_t)255; return p; };

    __bf16* WtQ = (__bf16*)take((size_t)DHEAD * DMODEL * 2);
    __bf16* WtK = (__bf16*)take((size_t)DHEAD * DMODEL * 2);
    __bf16* WtV = (__bf16*)take((size_t)DHEAD * DMODEL * 2);
    __bf16* Qb  = (__bf16*)take((size_t)BN * DHEAD * 2);
    __bf16* Kb  = (__bf16*)take((size_t)BN * DHEAD * 2);
    __bf16* Vt  = (__bf16*)take((size_t)BN * DHEAD * 2);   // [B][H][N]

    // 1) transpose+convert weights
    wt_convert<<<DMODEL, 128, 0, stream>>>(Wq, WtQ);
    wt_convert<<<DMODEL, 128, 0, stream>>>(Wk, WtK);
    wt_convert<<<DMODEL, 128, 0, stream>>>(Wv, WtV);

    // 2) projections (scale 1/sqrt(1024) folded into Q; V stored transposed)
    const int projGrid = (int)(BN / 64);           // 256 blocks
    proj_gemm<<<projGrid, 128, 0, stream>>>(X, WtQ, Qb, 0.03125f, 0);
    proj_gemm<<<projGrid, 128, 0, stream>>>(X, WtK, Kb, 1.0f,     0);
    proj_gemm<<<projGrid, 128, 0, stream>>>(X, WtV, Vt, 1.0f,     1);

    // 3) flash attention
    dim3 grid(NSEQ / 64, 4);
    flash_attn<<<grid, 128, 0, stream>>>(Qb, Kb, Vt, (float*)d_out);
}
